// GRUCell_62027917689552
// MI455X (gfx1250) — matrix operand order repaired and checked by reading, not compiled, not run on hardware
//
#include <hip/hip_runtime.h>
#include <hip/hip_bf16.h>

// GRU on MI455X (gfx1250): bf16 WMMA GEMMs with f32 accumulate.
//   Phase 0: convert+transpose weights f32[K][N] -> bf16[N][K] (workspace).
//   Phase 1: xg[g] = x @ wxh[g] + b[g]  (big parallel WMMA GEMM, 3 gates).
//   Phase 2: 512 sequential steps: (r,u) GEMMs -> sigmoid -> (r*h) GEMM ->
//            tanh -> h = u*c + (1-u)*h.  All launches on `stream`.
// Weight tiles are staged into LDS with GLOBAL_LOAD_ASYNC_TO_LDS_B128
// (ASYNCcnt-tracked, double-buffered), WMMA fragments read back via
// ds_load_b128.

typedef __bf16 bf16_t;
typedef __attribute__((ext_vector_type(16))) __bf16 v16bf;
typedef __attribute__((ext_vector_type(8)))  __bf16 bf16x8;
typedef __attribute__((ext_vector_type(8)))  float  v8f;
typedef __attribute__((ext_vector_type(4)))  float  f32x4;

#define B_  64
#define T_  512
#define D_  512
#define H_  1024
#define BT_ (B_ * T_)

// LDS B-tile: 64 columns x 32 K of bf16, row stride padded to 80 bytes
// (16B aligned for b128, spreads banks).
#define ROWB 40  // elements per LDS row (40 * 2B = 80B)

// ---------------------------------------------------------------------------
// CDNA5 async global->LDS helpers
// ---------------------------------------------------------------------------
typedef __attribute__((address_space(3))) void lds_void_t;

__device__ __forceinline__ unsigned lds_offset_of(void* p) {
  // ptrtoint of an AS(3) pointer yields the LDS byte offset, which is what
  // the async-load VDST operand expects (an offset added to the LDS base).
  return (unsigned)(unsigned long long)(lds_void_t*)p;
}

__device__ __forceinline__ void async_copy_b128(unsigned lds_off,
                                                const void* gptr) {
  asm volatile("global_load_async_to_lds_b128 %0, %1, off"
               :
               : "v"(lds_off), "v"((unsigned long long)gptr)
               : "memory");
}

__device__ __forceinline__ void wait_async_lds() {
  asm volatile("s_wait_asynccnt 0x0" ::: "memory");
}

// Stage a 64(N) x 32(K) bf16 tile of Wt[N][K] into LDS (4 KB) with async
// copies: 128 threads x 32B each (2 x b128 per thread).
__device__ __forceinline__ void stage_B_tile(const bf16_t* __restrict__ Wt,
                                             int ldK, int nBase, int kk,
                                             bf16_t* dst /* shared */) {
  const int t = threadIdx.x;        // 0..127
  const int n = t >> 1;             // row 0..63
  const int c = (t & 1) << 4;       // element chunk: 0 or 16
  const bf16_t* g = Wt + (long)(nBase + n) * ldK + kk + c;
  unsigned l = lds_offset_of(dst + n * ROWB + c);
  async_copy_b128(l, g);
  async_copy_b128(l + 16, (const char*)g + 16);
}

// ---------------------------------------------------------------------------
// Fragment loaders (CDNA5 WMMA 16-bit operand layouts, wave32)
// ---------------------------------------------------------------------------

// A-matrix 16x32 (MxK), f32 source converted to bf16.
// Lane L: row m = L%16, kb = (L/16)*8; VGPRs 0..3 hold K=kb..kb+7,
// VGPRs 4..7 hold K=kb+16..kb+23.
__device__ __forceinline__ v16bf load_fragA_f32(const float* __restrict__ A,
                                                int ld, int mBase, int kBase,
                                                int lane) {
  const int m  = mBase + (lane & 15);
  const int kb = kBase + ((lane >> 4) << 3);
  const float* p = A + (long)m * ld + kb;
  f32x4 a0 = *(const f32x4*)(p);
  f32x4 a1 = *(const f32x4*)(p + 4);
  f32x4 a2 = *(const f32x4*)(p + 16);
  f32x4 a3 = *(const f32x4*)(p + 20);
  v16bf f;
#pragma unroll
  for (int j = 0; j < 4; ++j) {
    f[j]      = (bf16_t)a0[j];
    f[4 + j]  = (bf16_t)a1[j];
    f[8 + j]  = (bf16_t)a2[j];
    f[12 + j] = (bf16_t)a3[j];
  }
  return f;
}

// B-matrix 32x16 (KxN) fragment from the LDS-staged [n][k] tile.
// Lane L: column n = nLocal0+L%16, kb = (L/16)*8; elements 0..7 hold
// K=kb..kb+7, elements 8..15 hold K=kb+16..kb+23 (two ds_load_b128 per lane).
__device__ __forceinline__ v16bf load_fragB_lds(const bf16_t* Bsh,
                                                int nLocal0, int lane) {
  const int n  = nLocal0 + (lane & 15);
  const int k0 = (lane >> 4) << 3;          // 0 or 8
  const bf16_t* p = Bsh + n * ROWB + k0;
  bf16x8 b0 = *(const bf16x8*)(p);
  bf16x8 b1 = *(const bf16x8*)(p + 16);
  v16bf f;
#pragma unroll
  for (int j = 0; j < 8; ++j) {
    f[j]     = b0[j];
    f[8 + j] = b1[j];
  }
  return f;
}

__device__ __forceinline__ v8f wmma_bf16(v16bf a, v16bf b, v8f c) {
  // (neg_a, A, neg_b, B, c_mod, C, reuse_a, reuse_b)
  return __builtin_amdgcn_wmma_f32_16x16x32_bf16(false, a, false, b,
                                                 (short)0, c, false, false);
}

// ---------------------------------------------------------------------------
// Phase 0: weight convert + transpose  f32[K][N] -> bf16[N][K]
// ---------------------------------------------------------------------------
__global__ void convert_transpose(const float* __restrict__ W,
                                  bf16_t* __restrict__ Wt, int K, int N) {
  long idx = (long)blockIdx.x * blockDim.x + threadIdx.x;
  if (idx >= (long)K * N) return;
  int k = (int)(idx / N);
  int n = (int)(idx % N);
  Wt[(long)n * K + k] = (bf16_t)W[(long)k * N + n];
}

// ---------------------------------------------------------------------------
// Phase 1: xg[g][row][h] = x[row] @ wxh[g] + b[g],  row = b*T + t
// grid = (H/64, BT/64, 3), block = 128 (4 waves); wave w -> rows mBase..+15
// ---------------------------------------------------------------------------
__global__ void __launch_bounds__(128)
xproj_gemm(const float* __restrict__ x, const bf16_t* __restrict__ wxh_t,
           const float* __restrict__ rb, const float* __restrict__ ub,
           const float* __restrict__ cb, float* __restrict__ xg) {
  __shared__ bf16_t Bsh[2][64 * ROWB];
  const int lane  = threadIdx.x & 31;
  const int wave  = threadIdx.x >> 5;
  const int g     = blockIdx.z;
  const int nBase = blockIdx.x * 64;
  const int mBase = blockIdx.y * 64 + wave * 16;
  const bf16_t* Wt = wxh_t + (long)g * D_ * H_;  // [H_][D_]

  stage_B_tile(Wt, D_, nBase, 0, Bsh[0]);
  wait_async_lds();
  __syncthreads();

  v8f acc[4] = {};
  for (int kk = 0; kk < D_; kk += 32) {
    const int cur = (kk >> 5) & 1;
    if (kk + 32 < D_)  // start async copy of next K-slab (double buffer)
      stage_B_tile(Wt, D_, nBase, kk + 32, Bsh[cur ^ 1]);
    __builtin_prefetch(x + (long)(mBase + (lane & 15)) * D_ + kk + 32, 0, 1);
    v16bf a = load_fragA_f32(x, D_, mBase, kk, lane);
#pragma unroll
    for (int j = 0; j < 4; ++j) {
      v16bf b = load_fragB_lds(Bsh[cur], j * 16, lane);
      acc[j] = wmma_bf16(a, b, acc[j]);
    }
    wait_async_lds();
    __syncthreads();
  }

  const float* bias = (g == 0) ? rb : (g == 1) ? ub : cb;
  float* outg = xg + (long)g * BT_ * H_;
#pragma unroll
  for (int j = 0; j < 4; ++j)
#pragma unroll
    for (int r = 0; r < 8; ++r) {
      int row = mBase + ((lane >> 4) << 3) + r;
      int col = nBase + j * 16 + (lane & 15);
      outg[(long)row * H_ + col] = acc[j][r] + bias[col];
    }
}

// ---------------------------------------------------------------------------
// Phase 2a: r and u gates for timestep t.
// grid = (H/64, 2): blockIdx.y == 0 -> r (also writes r*h), 1 -> u.
// ---------------------------------------------------------------------------
__global__ void __launch_bounds__(128)
step_ru(const float* __restrict__ hbuf, const bf16_t* __restrict__ whh_t,
        const float* __restrict__ xg, int t, float* __restrict__ rh,
        float* __restrict__ ubuf) {
  __shared__ bf16_t Bsh[2][64 * ROWB];
  const int lane  = threadIdx.x & 31;
  const int wave  = threadIdx.x >> 5;
  const int g     = blockIdx.y;          // 0 = relevance, 1 = update
  const int nBase = blockIdx.x * 64;
  const int mBase = wave * 16;           // 4 waves cover B_ = 64 rows
  const bf16_t* Wt = whh_t + (long)g * H_ * H_;  // [H_][H_] transposed

  stage_B_tile(Wt, H_, nBase, 0, Bsh[0]);
  wait_async_lds();
  __syncthreads();

  v8f acc[4] = {};
  for (int kk = 0; kk < H_; kk += 32) {
    const int cur = (kk >> 5) & 1;
    if (kk + 32 < H_)
      stage_B_tile(Wt, H_, nBase, kk + 32, Bsh[cur ^ 1]);
    __builtin_prefetch(hbuf + (long)(mBase + (lane & 15)) * H_ + kk + 32, 0, 1);
    v16bf a = load_fragA_f32(hbuf, H_, mBase, kk, lane);
#pragma unroll
    for (int j = 0; j < 4; ++j) {
      v16bf b = load_fragB_lds(Bsh[cur], j * 16, lane);
      acc[j] = wmma_bf16(a, b, acc[j]);
    }
    wait_async_lds();
    __syncthreads();
  }

  const float* xgt = xg + (long)g * BT_ * H_;
#pragma unroll
  for (int j = 0; j < 4; ++j)
#pragma unroll
    for (int r = 0; r < 8; ++r) {
      int m   = mBase + ((lane >> 4) << 3) + r;   // batch index
      int col = nBase + j * 16 + (lane & 15);
      float v = acc[j][r] + xgt[((long)m * T_ + t) * H_ + col];
      float s = 1.0f / (1.0f + __expf(-v));
      if (g == 0)
        rh[m * H_ + col] = s * hbuf[m * H_ + col];
      else
        ubuf[m * H_ + col] = s;
    }
}

// ---------------------------------------------------------------------------
// Phase 2b: candidate + state update for timestep t.  grid = (H/64)
// ---------------------------------------------------------------------------
__global__ void __launch_bounds__(128)
step_c(const float* __restrict__ rh, const bf16_t* __restrict__ whhc_t,
       const float* __restrict__ xg, int t, const float* __restrict__ ubuf,
       float* __restrict__ hbuf, float* __restrict__ out) {
  __shared__ bf16_t Bsh[2][64 * ROWB];
  const int lane  = threadIdx.x & 31;
  const int wave  = threadIdx.x >> 5;
  const int nBase = blockIdx.x * 64;
  const int mBase = wave * 16;

  stage_B_tile(whhc_t, H_, nBase, 0, Bsh[0]);
  wait_async_lds();
  __syncthreads();

  v8f acc[4] = {};
  for (int kk = 0; kk < H_; kk += 32) {
    const int cur = (kk >> 5) & 1;
    if (kk + 32 < H_)
      stage_B_tile(whhc_t, H_, nBase, kk + 32, Bsh[cur ^ 1]);
    __builtin_prefetch(rh + (long)(mBase + (lane & 15)) * H_ + kk + 32, 0, 1);
    v16bf a = load_fragA_f32(rh, H_, mBase, kk, lane);
#pragma unroll
    for (int j = 0; j < 4; ++j) {
      v16bf b = load_fragB_lds(Bsh[cur], j * 16, lane);
      acc[j] = wmma_bf16(a, b, acc[j]);
    }
    wait_async_lds();
    __syncthreads();
  }

  const float* xct = xg + (long)2 * BT_ * H_;
#pragma unroll
  for (int j = 0; j < 4; ++j)
#pragma unroll
    for (int r = 0; r < 8; ++r) {
      int m   = mBase + ((lane >> 4) << 3) + r;
      int col = nBase + j * 16 + (lane & 15);
      float c  = tanhf(acc[j][r] + xct[((long)m * T_ + t) * H_ + col]);
      float u  = ubuf[m * H_ + col];
      float ho = hbuf[m * H_ + col];
      float hn = u * c + (1.0f - u) * ho;
      hbuf[m * H_ + col] = hn;
      out[((long)m * T_ + t) * H_ + col] = hn;   // output layout [B,T,H]
    }
}

// ---------------------------------------------------------------------------
// Host side
// ---------------------------------------------------------------------------
extern "C" void kernel_launch(void* const* d_in, const int* in_sizes, int n_in,
                              void* d_out, int out_size, void* d_ws,
                              size_t ws_size, hipStream_t stream) {
  (void)in_sizes; (void)n_in; (void)out_size; (void)ws_size;

  const float* x    = (const float*)d_in[0];
  const float* h0   = (const float*)d_in[1];
  const float* rwhh = (const float*)d_in[2];
  const float* rwxh = (const float*)d_in[3];
  const float* rb   = (const float*)d_in[4];
  const float* uwhh = (const float*)d_in[5];
  const float* uwxh = (const float*)d_in[6];
  const float* ub   = (const float*)d_in[7];
  const float* cwhh = (const float*)d_in[8];
  const float* cwxh = (const float*)d_in[9];
  const float* cb   = (const float*)d_in[10];
  float* out = (float*)d_out;

  char* ws = (char*)d_ws;
  size_t off = 0;
  auto walloc = [&](size_t bytes) -> void* {
    off = (off + 255) & ~(size_t)255;
    void* p = ws + off;
    off += bytes;
    return p;
  };
  const long nDH = (long)D_ * H_;
  const long nHH = (long)H_ * H_;
  bf16_t* wxh_t = (bf16_t*)walloc(3 * (size_t)nDH * sizeof(bf16_t)); //  3 MB
  bf16_t* whh_t = (bf16_t*)walloc(3 * (size_t)nHH * sizeof(bf16_t)); //  6 MB
  float*  hbuf  = (float*)walloc((size_t)B_ * H_ * sizeof(float));
  float*  ubuf  = (float*)walloc((size_t)B_ * H_ * sizeof(float));
  float*  rhbuf = (float*)walloc((size_t)B_ * H_ * sizeof(float));
  float*  xg    = (float*)walloc(3 * (size_t)BT_ * H_ * sizeof(float)); // 384 MB

  // Phase 0: weight convert + transpose (one-time, tiny).
  const int thr = 256;
  convert_transpose<<<(nDH + thr - 1) / thr, thr, 0, stream>>>(rwxh, wxh_t + 0 * nDH, D_, H_);
  convert_transpose<<<(nDH + thr - 1) / thr, thr, 0, stream>>>(uwxh, wxh_t + 1 * nDH, D_, H_);
  convert_transpose<<<(nDH + thr - 1) / thr, thr, 0, stream>>>(cwxh, wxh_t + 2 * nDH, D_, H_);
  convert_transpose<<<(nHH + thr - 1) / thr, thr, 0, stream>>>(rwhh, whh_t + 0 * nHH, H_, H_);
  convert_transpose<<<(nHH + thr - 1) / thr, thr, 0, stream>>>(uwhh, whh_t + 1 * nHH, H_, H_);
  convert_transpose<<<(nHH + thr - 1) / thr, thr, 0, stream>>>(cwhh, whh_t + 2 * nHH, H_, H_);

  // h state (never mutate d_in).
  hipMemcpyAsync(hbuf, h0, (size_t)B_ * H_ * sizeof(float),
                 hipMemcpyDeviceToDevice, stream);

  // Phase 1: input projections for all timesteps / gates.
  xproj_gemm<<<dim3(H_ / 64, BT_ / 64, 3), 128, 0, stream>>>(x, wxh_t, rb, ub,
                                                             cb, xg);

  // Phase 2: sequential scan over time.
  for (int t = 0; t < T_; ++t) {
    step_ru<<<dim3(H_ / 64, 2), 128, 0, stream>>>(hbuf, whh_t, xg, t, rhbuf,
                                                  ubuf);
    step_c<<<dim3(H_ / 64, 1), 128, 0, stream>>>(rhbuf, whh_t + 2 * nHH, xg, t,
                                                 ubuf, hbuf, out);
  }
}
